// MessagePassing_65214783423009
// MI455X (gfx1250) — matrix operand order repaired and checked by reading, not compiled, not run on hardware
//
#include <hip/hip_runtime.h>
#include <hip/hip_bf16.h>

// ---------------------------------------------------------------------------
// MPNN (Gilmer) for MI455X / gfx1250, wave32 + WMMA bf16.
//   N=40000 nodes, E=640000 edges, H=16, 8 passes, 64 graphs.
// Strategy (roofline): dominant object is the per-edge 16x16 message matrix
// a[E,16,16] (bf16: 327MB). Built ONCE with v_wmma_f32_16x16x32_bf16
// (84 GFLOP), then streamed 8x (~3GB @23.3TB/s ~ 128us floor). bvec's
// segment-sum is pass-invariant and hoisted. h/m buffers (2.5MB) and all
// weights stay L2 resident. Readout GEMMs ([N,256]@[256,256] x2) also WMMA.
// ---------------------------------------------------------------------------

#define NNODES 40000
#define NEDGES 640000
#define NGRAPHS 64
#define NH 16

typedef __bf16 bf16_t;
typedef bf16_t v16bf __attribute__((ext_vector_type(16)));
typedef float  v8f   __attribute__((ext_vector_type(8)));

__device__ __forceinline__ float selu_f(float x) {
  const float scale = 1.0507009873554805f;
  const float alpha = 1.6732632423543772f;
  return x > 0.f ? scale * x : scale * alpha * (__expf(x) - 1.f);
}
__device__ __forceinline__ float sigmoid_f(float x) {
  return 1.f / (1.f + __expf(-x));
}

// ---------------------------------------------------------------------------
// Repack a row-major f32 weight matrix [256, ncols] into the wave32 WMMA-B
// fragment layout, bf16:  dst[((nt*8+kb)*32+lane)*16 + t]
//   column n = nt*16 + (lane&15), k = kb*32 + (lane>>4)*8 + (t<8 ? t : t+8)
// so each lane's 16 K-values for its column are one contiguous 32B load.
// ---------------------------------------------------------------------------
__global__ __launch_bounds__(256)
void repack_b_kernel(const float* __restrict__ src, bf16_t* __restrict__ dst,
                     int ncols) {
  int tid = blockIdx.x * blockDim.x + threadIdx.x;
  int total = 256 * ncols;
  if (tid >= total) return;
  int t    = tid & 15;
  int lane = (tid >> 4) & 31;
  int kb   = (tid >> 9) & 7;
  int nt   = tid >> 12;
  int n = nt * 16 + (lane & 15);
  int k = kb * 32 + (lane >> 4) * 8 + (t < 8 ? t : t + 8);
  dst[tid] = (bf16_t)src[k * ncols + n];
}

// h0 = pad(x), zero m/c accumulators and graph accumulator.
__global__ __launch_bounds__(256)
void init_kernel(const float* __restrict__ x, float* __restrict__ h,
                 float* __restrict__ mbuf, float* __restrict__ cbuf,
                 float* __restrict__ gsum) {
  int idx = blockIdx.x * blockDim.x + threadIdx.x;
  if (idx < NNODES * NH) {
    int n = idx >> 4, i = idx & 15;
    h[idx] = (i < 2) ? x[n * 2 + i] : 0.f;
    mbuf[idx] = 0.f;
    cbuf[idx] = 0.f;
  }
  if (idx < NGRAPHS * 256) gsum[idx] = 0.f;
}

// ---------------------------------------------------------------------------
// Edge precompute: one wave per 16-edge tile.
//  s  = selu(e*l_w1+l_b1)  -> bf16 WMMA-A frags (computed in-lane)
//  a  = s @ l_w2 + l_b2    -> 16 col-tiles x 8 K-steps of WMMA, stored bf16
//  bv = selu(e*b_w1+b_b1) @ b_w2 + b_b2, scatter segment-summed into cbuf
//      (pass-invariant, so it never appears in the 8-pass loop).
// ---------------------------------------------------------------------------
__global__ __launch_bounds__(256)
void edge_weights_kernel(const float* __restrict__ ef,
                         const int* __restrict__ second,
                         const float* __restrict__ lw1, const float* __restrict__ lb1,
                         const float* __restrict__ lb2,
                         const float* __restrict__ bw1, const float* __restrict__ bb1,
                         const float* __restrict__ bb2,
                         const v16bf* __restrict__ w2b,   // [16][8][32] lane-blobs
                         const v16bf* __restrict__ b2b,   // [8][32]
                         bf16_t* __restrict__ aBuf,
                         float* __restrict__ cbuf) {
  const int wave = threadIdx.x >> 5;
  const int lane = threadIdx.x & 31;
  const int eb = blockIdx.x * 8 + wave;
  if (eb >= NEDGES / 16) return;
  const int e0 = eb * 16;
  const int lm = lane & 15;
  const int lh = lane >> 4;
  const float ev = ef[e0 + lm];  // edge feature of this lane's A-row

  // ---- A fragments for the l-path (16x32 bf16 A layout) ----
  v16bf aF[8];
#pragma unroll
  for (int kb = 0; kb < 8; ++kb) {
#pragma unroll
    for (int t = 0; t < 16; ++t) {
      int k = kb * 32 + lh * 8 + (t < 8 ? t : t + 8);
      aF[kb][t] = (bf16_t)selu_f(ev * lw1[k] + lb1[k]);
    }
  }
  // ---- a = s @ l_w2 + l_b2, stored bf16 row-major [E,256] ----
#pragma unroll 1
  for (int nt = 0; nt < 16; ++nt) {
    v8f c = {};
#pragma unroll
    for (int kb = 0; kb < 8; ++kb) {
      v16bf bF = w2b[(nt * 8 + kb) * 32 + lane];
      c = __builtin_amdgcn_wmma_f32_16x16x32_bf16(false, aF[kb], false, bF,
                                                  (short)0, c, false, false);
    }
    int col = nt * 16 + lm;
    float bias = lb2[col];
#pragma unroll
    for (int r = 0; r < 8; ++r) {
      int mrow = r + lh * 8;
      aBuf[(size_t)(e0 + mrow) * 256 + col] = (bf16_t)(c[r] + bias);
    }
  }
  // ---- bvec path + hoisted segment-sum into cbuf ----
  v16bf sbF[8];
#pragma unroll
  for (int kb = 0; kb < 8; ++kb) {
#pragma unroll
    for (int t = 0; t < 16; ++t) {
      int k = kb * 32 + lh * 8 + (t < 8 ? t : t + 8);
      sbF[kb][t] = (bf16_t)selu_f(ev * bw1[k] + bb1[k]);
    }
  }
  v8f cb = {};
#pragma unroll
  for (int kb = 0; kb < 8; ++kb) {
    v16bf bF = b2b[kb * 32 + lane];
    cb = __builtin_amdgcn_wmma_f32_16x16x32_bf16(false, sbF[kb], false, bF,
                                                 (short)0, cb, false, false);
  }
  float bias = bb2[lm];
#pragma unroll
  for (int r = 0; r < 8; ++r) {
    int mrow = r + lh * 8;
    atomicAdd(&cbuf[second[e0 + mrow] * NH + lm], cb[r] + bias);
  }
}

// ---------------------------------------------------------------------------
// Message pass: one lane per edge. Stream the edge's 512B bf16 matrix row,
// gather h[first] (L2 hit), 16 f32 atomics into L2-resident m buffer.
// ---------------------------------------------------------------------------
__global__ __launch_bounds__(256)
void message_kernel(const bf16_t* __restrict__ aBuf,
                    const float* __restrict__ h,
                    const int* __restrict__ first,
                    const int* __restrict__ second,
                    float* __restrict__ mbuf) {
  int e = blockIdx.x * blockDim.x + threadIdx.x;
  if (e >= NEDGES) return;
  const bf16_t* arow = aBuf + (size_t)e * 256;
  if (e + 4096 < NEDGES)  // stream-ahead hint -> global_prefetch_b8
    __builtin_prefetch(aBuf + (size_t)(e + 4096) * 256, 0, 0);
  int f = first[e];
  float hf[16];
#pragma unroll
  for (int j = 0; j < 16; ++j) hf[j] = h[f * NH + j];
  int base = second[e] * NH;
#pragma unroll
  for (int i = 0; i < 16; ++i) {
    v16bf av = *(const v16bf*)(arow + i * 16);
    float s = 0.f;
#pragma unroll
    for (int j = 0; j < 16; ++j) s += (float)av[j] * hf[j];
    atomicAdd(&mbuf[base + i], s);
  }
}

// ---------------------------------------------------------------------------
// Keras GRUCell (reset_after=True). One thread per node; weight indices are
// wave-uniform so K/RK reads go through the scalar path. Re-zeros mbuf.
// ---------------------------------------------------------------------------
__global__ __launch_bounds__(256)
void gru_kernel(float* __restrict__ h, float* __restrict__ mbuf,
                const float* __restrict__ cbuf,
                const float* __restrict__ K, const float* __restrict__ RK,
                const float* __restrict__ gb) {
  int n = blockIdx.x * blockDim.x + threadIdx.x;
  if (n >= NNODES) return;
  float m[16], hv[16];
#pragma unroll
  for (int i = 0; i < 16; ++i) {
    m[i] = mbuf[n * NH + i] + cbuf[n * NH + i];
    hv[i] = h[n * NH + i];
  }
  float out[16];
#pragma unroll
  for (int c = 0; c < 16; ++c) {
    float xz = gb[c], xr = gb[16 + c], xh = gb[32 + c];
    float rz = gb[48 + c], rr = gb[64 + c], rh = gb[80 + c];
#pragma unroll
    for (int j = 0; j < 16; ++j) {
      xz += m[j] * K[j * 48 + c];
      xr += m[j] * K[j * 48 + 16 + c];
      xh += m[j] * K[j * 48 + 32 + c];
      rz += hv[j] * RK[j * 48 + c];
      rr += hv[j] * RK[j * 48 + 16 + c];
      rh += hv[j] * RK[j * 48 + 32 + c];
    }
    float z = sigmoid_f(xz + rz);
    float r = sigmoid_f(xr + rr);
    float hh = tanhf(xh + r * rh);
    out[c] = z * hv[c] + (1.f - z) * hh;
  }
#pragma unroll
  for (int i = 0; i < 16; ++i) {
    h[n * NH + i] = out[i];
    mbuf[n * NH + i] = 0.f;  // ready for next pass
  }
}

// ---------------------------------------------------------------------------
// Readout: one wave per 16-node tile. Stage 1 (K=18 layers + tanh/selu) is
// computed cooperatively into LDS as bf16, then WMMA A-frags are built from
// LDS and the two 256x256 GEMMs run as WMMA; gate*feat scatter-adds into the
// 64x256 graph accumulator.
// ---------------------------------------------------------------------------
__global__ __launch_bounds__(32)
void readout_kernel(const float* __restrict__ h, const float* __restrict__ x,
                    const int* __restrict__ segment,
                    const float* __restrict__ iw1, const float* __restrict__ ib1,
                    const float* __restrict__ ib2,
                    const float* __restrict__ jw1, const float* __restrict__ jb1,
                    const float* __restrict__ jb2,
                    const v16bf* __restrict__ iw2b, const v16bf* __restrict__ jw2b,
                    float* __restrict__ gsum) {
  __shared__ float  s_hx[16][18];
  __shared__ bf16_t s_t1[16 * 256];
  __shared__ bf16_t s_t2[16 * 256];

  const int lane = threadIdx.x & 31;
  const int nb = blockIdx.x;
  if (nb >= NNODES / 16) return;
  const int n0 = nb * 16;
  const int lm = lane & 15;
  const int lh = lane >> 4;

  // stage hx = [h | x] into LDS
  for (int idx = lane; idx < 16 * 18; idx += 32) {
    int m = idx / 18, j = idx % 18;
    s_hx[m][j] = (j < 16) ? h[(n0 + m) * NH + j] : x[(n0 + m) * 2 + (j - 16)];
  }
  __syncthreads();

  // stage 1: t1 = tanh(hx@i_w1+i_b1), t2 = selu(hx@j_w1+j_b1), bf16 in LDS
  for (int idx = lane; idx < 16 * 256; idx += 32) {
    int m = idx >> 8, k = idx & 255;
    float s1 = ib1[k], s2 = jb1[k];
#pragma unroll
    for (int j = 0; j < 18; ++j) {
      float hx = s_hx[m][j];
      s1 += hx * iw1[j * 256 + k];
      s2 += hx * jw1[j * 256 + k];
    }
    s_t1[idx] = (bf16_t)tanhf(s1);
    s_t2[idx] = (bf16_t)selu_f(s2);
  }
  __syncthreads();

  // build WMMA A fragments from LDS (16x32 bf16 A layout)
  v16bf t1F[8], t2F[8];
#pragma unroll
  for (int kb = 0; kb < 8; ++kb) {
#pragma unroll
    for (int t = 0; t < 16; ++t) {
      int k = kb * 32 + lh * 8 + (t < 8 ? t : t + 8);
      t1F[kb][t] = s_t1[lm * 256 + k];
      t2F[kb][t] = s_t2[lm * 256 + k];
    }
  }
  int segr[8];
#pragma unroll
  for (int r = 0; r < 8; ++r) segr[r] = segment[n0 + r + lh * 8];

#pragma unroll 1
  for (int nt = 0; nt < 16; ++nt) {
    v8f cg = {}, cf = {};
#pragma unroll
    for (int kb = 0; kb < 8; ++kb) {
      v16bf bi = iw2b[(nt * 8 + kb) * 32 + lane];
      cg = __builtin_amdgcn_wmma_f32_16x16x32_bf16(false, t1F[kb], false, bi,
                                                   (short)0, cg, false, false);
      v16bf bj = jw2b[(nt * 8 + kb) * 32 + lane];
      cf = __builtin_amdgcn_wmma_f32_16x16x32_bf16(false, t2F[kb], false, bj,
                                                   (short)0, cf, false, false);
    }
    int col = nt * 16 + lm;
    float bg = ib2[col], bfv = jb2[col];
#pragma unroll
    for (int r = 0; r < 8; ++r) {
      float gate = sigmoid_f(cg[r] + bg);
      float feat = cf[r] + bfv;
      atomicAdd(&gsum[segr[r] * 256 + col], gate * feat);
    }
  }
}

// Final head: out[g] = selu(gsum[g]@f_w1+f_b1) @ f_w2 + f_b2. 64x256x256.
__global__ __launch_bounds__(256)
void final_kernel(const float* __restrict__ gsum,
                  const float* __restrict__ fw1, const float* __restrict__ fb1,
                  const float* __restrict__ fw2, const float* __restrict__ fb2,
                  float* __restrict__ out) {
  __shared__ float red[256];
  int g = blockIdx.x, t = threadIdx.x;
  const float* grow = gsum + g * 256;  // row reads are wave-uniform -> SMEM
  float s = fb1[t];
#pragma unroll 4
  for (int k = 0; k < 256; ++k) s += grow[k] * fw1[k * 256 + t];
  red[t] = selu_f(s) * fw2[t];
  __syncthreads();
  for (int st = 128; st > 0; st >>= 1) {
    if (t < st) red[t] += red[t + st];
    __syncthreads();
  }
  if (t == 0) out[g] = red[0] + fb2[0];
}

// ---------------------------------------------------------------------------
extern "C" void kernel_launch(void* const* d_in, const int* in_sizes, int n_in,
                              void* d_out, int out_size, void* d_ws,
                              size_t ws_size, hipStream_t stream) {
  (void)in_sizes; (void)n_in; (void)out_size; (void)ws_size;
  const float* x      = (const float*)d_in[0];
  const float* e      = (const float*)d_in[1];
  const int*   first  = (const int*)d_in[2];
  const int*   second = (const int*)d_in[3];
  const int*   segment= (const int*)d_in[4];
  const float* l_w1 = (const float*)d_in[5];
  const float* l_b1 = (const float*)d_in[6];
  const float* l_w2 = (const float*)d_in[7];
  const float* l_b2 = (const float*)d_in[8];
  const float* b_w1 = (const float*)d_in[9];
  const float* b_b1 = (const float*)d_in[10];
  const float* b_w2 = (const float*)d_in[11];
  const float* b_b2 = (const float*)d_in[12];
  const float* gru_k  = (const float*)d_in[13];
  const float* gru_rk = (const float*)d_in[14];
  const float* gru_b  = (const float*)d_in[15];
  const float* i_w1 = (const float*)d_in[16];
  const float* i_b1 = (const float*)d_in[17];
  const float* i_w2 = (const float*)d_in[18];
  const float* i_b2 = (const float*)d_in[19];
  const float* j_w1 = (const float*)d_in[20];
  const float* j_b1 = (const float*)d_in[21];
  const float* j_w2 = (const float*)d_in[22];
  const float* j_b2 = (const float*)d_in[23];
  const float* f_w1 = (const float*)d_in[24];
  const float* f_b1 = (const float*)d_in[25];
  const float* f_w2 = (const float*)d_in[26];
  const float* f_b2 = (const float*)d_in[27];

  // workspace carve (256B aligned); dominant: aBuf bf16 [E,256] = 327.7MB
  char* ws = (char*)d_ws;
  size_t off = 0;
  auto take = [&](size_t bytes) -> void* {
    void* p = ws + off;
    off = (off + bytes + 255) & ~(size_t)255;
    return p;
  };
  bf16_t* aBuf = (bf16_t*)take((size_t)NEDGES * 256 * sizeof(bf16_t));
  v16bf* w2b  = (v16bf*)take(256 * 256 * sizeof(bf16_t));
  v16bf* b2b  = (v16bf*)take(256 * 16 * sizeof(bf16_t));
  v16bf* iw2b = (v16bf*)take(256 * 256 * sizeof(bf16_t));
  v16bf* jw2b = (v16bf*)take(256 * 256 * sizeof(bf16_t));
  float* hbuf = (float*)take((size_t)NNODES * NH * sizeof(float));
  float* mbuf = (float*)take((size_t)NNODES * NH * sizeof(float));
  float* cbuf = (float*)take((size_t)NNODES * NH * sizeof(float));
  float* gsum = (float*)take((size_t)NGRAPHS * 256 * sizeof(float));

  // repack B matrices into WMMA-B lane layout (bf16)
  repack_b_kernel<<<(256 * 256 + 255) / 256, 256, 0, stream>>>(l_w2, (bf16_t*)w2b, 256);
  repack_b_kernel<<<(256 * 16 + 255) / 256, 256, 0, stream>>>(b_w2, (bf16_t*)b2b, 16);
  repack_b_kernel<<<(256 * 256 + 255) / 256, 256, 0, stream>>>(i_w2, (bf16_t*)iw2b, 256);
  repack_b_kernel<<<(256 * 256 + 255) / 256, 256, 0, stream>>>(j_w2, (bf16_t*)jw2b, 256);

  // h0 = pad(x); zero m, c, gsum
  init_kernel<<<(NNODES * NH + 255) / 256, 256, 0, stream>>>(x, hbuf, mbuf, cbuf, gsum);

  // edge-conditioned weights (once) + hoisted bvec segment-sum
  edge_weights_kernel<<<NEDGES / 16 / 8, 256, 0, stream>>>(
      e, second, l_w1, l_b1, l_b2, b_w1, b_b1, b_b2, w2b, b2b, aBuf, cbuf);

  // 8 message-passing rounds
  for (int p = 0; p < 8; ++p) {
    message_kernel<<<NEDGES / 256, 256, 0, stream>>>(aBuf, hbuf, first, second, mbuf);
    gru_kernel<<<(NNODES + 255) / 256, 256, 0, stream>>>(hbuf, mbuf, cbuf,
                                                         gru_k, gru_rk, gru_b);
  }

  // gated readout + final head
  readout_kernel<<<NNODES / 16, 32, 0, stream>>>(hbuf, x, segment,
                                                 i_w1, i_b1, i_b2,
                                                 j_w1, j_b1, j_b2,
                                                 iw2b, jw2b, gsum);
  final_kernel<<<NGRAPHS, 256, 0, stream>>>(gsum, f_w1, f_b1, f_w2, f_b2,
                                            (float*)d_out);
}
